// RetNet_16535624089815
// MI455X (gfx1250) — compile-verified, matrix operand order checked by reading, not run on hardware
//
#include <hip/hip_runtime.h>
#include <hip/hip_bf16.h>
#include <math.h>

// ---------------------------------------------------------------------------
// MI455X (gfx1250) implementation. wave32, WMMA f32_16x16x32_f16 everywhere.
// One workgroup (4 waves / 128 threads) per frame; all activations in LDS.
// Weights packed once per launch into WMMA-B-fragment tiled f16 layout.
// Global reads use explicit AS(1) casts so they lower to global_load_* (not
// flat_load_*, which would also tie up DScnt / the LDS path on CDNA5).
// ---------------------------------------------------------------------------

typedef _Float16 half_t;
typedef __attribute__((ext_vector_type(16))) _Float16 v16h;
typedef __attribute__((ext_vector_type(8)))  _Float16 v8h;
typedef __attribute__((ext_vector_type(8)))  float    v8f;

// global-address-space views (lower to global_load_b128 / global_load_b32)
typedef const __attribute__((address_space(1))) v8h    gv8h_t;
typedef const __attribute__((address_space(1))) float  gfloat_t;
typedef const __attribute__((address_space(1))) half_t ghalf_t;

#define NWAVES 4
#define BLOCK  128
#define JN     22
#define NPAD   32

__device__ __forceinline__ float gload_f(const float* p) {
    return *(gfloat_t*)p;
}

// ---------------- fragment loaders ----------------
// ISA 16-bit A-matrix 16x32 layout: lane (hi=lane>>4, r=lane&15) holds row M=r,
// halves[0..7]  = A[r][8*hi + 0..7]          (contiguous, 16B -> ds_load_b128)
// halves[8..15] = A[r][16 + 8*hi + 0..7]     (contiguous, 16B -> ds_load_b128)
// B operand (K x 16) uses the same striping with n = lane&15, so a B^T stored
// row-major (rows indexed by n) loads with the identical gather.
__device__ __forceinline__ v16h load_frag_lds(const half_t* base, int ld) {
    const int lane = threadIdx.x & 31;
    const int hi = lane >> 4, r = lane & 15;
    const half_t* p = base + r * ld + (hi << 3);
    union { v16h v; v8h g[2]; } u;
    u.g[0] = *(const v8h*)(p);
    u.g[1] = *(const v8h*)(p + 16);
    return u.v;
}

// masked variant for K-valid < 32 (hd=16 attention heads)
__device__ __forceinline__ v16h load_frag_lds_kv(const half_t* base, int ld, int kvalid) {
    const int lane = threadIdx.x & 31;
    const int hi = lane >> 4, r = lane & 15;
    const half_t* p = base + r * ld + (hi << 3);
    union { v16h v; v8h g[2]; } u;
    v8h z = {};
    u.g[0] = ((hi << 3) < kvalid)        ? *(const v8h*)(p)      : z;
    u.g[1] = ((16 + (hi << 3)) < kvalid) ? *(const v8h*)(p + 16) : z;
    return u.v;
}

// packed weight tile: 512 halves per 32x16 tile, lane-major (lane*16 + j)
// -> two global_load_b128 per lane
__device__ __forceinline__ v16h load_frag_w(const half_t* p) {
    union { v16h v; v8h g[2]; } u;
    u.g[0] = *(gv8h_t*)(p);
    u.g[1] = *(gv8h_t*)(p + 8);
    return u.v;
}

// ---------------- generic GEMM: out(32 x N) = A(32 x K) @ Wt + bias ----------
template <bool RELU, bool RESID>
__device__ __forceinline__ void mm_w(const half_t* A, int lda,
                                     const half_t* Wt, const float* bias,
                                     int K, int N,
                                     half_t* out, int ldo, int out_col,
                                     const half_t* resid, int ldr) {
    const int ntk  = K >> 5;
    const int ntn  = N >> 4;
    const int wave = threadIdx.x >> 5;
    const int lane = threadIdx.x & 31;
    const int hi = lane >> 4, nl = lane & 15;
    for (int t = wave; t < 2 * ntn; t += NWAVES) {
        const int mt = t & 1, nt = t >> 1;
        v8f acc = {};
        const half_t* wp = Wt + (((size_t)nt * ntk) << 9) + (lane << 4);
        const half_t* ap = A + mt * 16 * lda;
#pragma unroll 2
        for (int kt = 0; kt < ntk; ++kt) {
            if (kt + 1 < ntk)
                __builtin_prefetch((const void*)(wp + ((kt + 1) << 9)), 0, 3);
            v16h a = load_frag_lds(ap + (kt << 5), lda);
            v16h b = load_frag_w(wp + (kt << 9));
            acc = __builtin_amdgcn_wmma_f32_16x16x32_f16(false, a, false, b,
                                                         (short)0, acc, false, false);
        }
        const int n = (nt << 4) + nl;
        float bv = gload_f(bias + n);
        union { v8f v; float f[8]; } u; u.v = acc;
#pragma unroll
        for (int r = 0; r < 8; ++r) {
            int m = (mt << 4) + (hi << 3) + r;
            float v = u.f[r] + bv;
            if (RESID) v += (float)resid[m * ldr + n];
            if (RELU)  v = fmaxf(v, 0.0f);
            out[m * ldo + out_col + n] = (half_t)v;
        }
    }
}

// ---------------- encoder (shared by Quat/Skel encoders) --------------------
__device__ void encoder(const float* x, int in_dim, int heads, int hd, int d_o,
                        const half_t* Wtok, const float* btok,
                        const half_t* Wqkv, const float* bqkv,
                        const half_t* Wo,   const float* bo,
                        const float* ln_g,  const float* ln_b,
                        const half_t* Wm1,  const float* bm1,
                        const half_t* Wm2,  const float* bm2,
                        int cat_ofs,
                        half_t* s_x, half_t* s_tok, half_t* s_qkv, half_t* s_vT,
                        float* s_dots, half_t* s_attn, half_t* s_ctx,
                        half_t* s_h, half_t* s_hn, half_t* s_mlp, half_t* s_cat) {
    const int tid  = threadIdx.x;
    const int wave = tid >> 5;
    const int lane = tid & 31;
    const int hi = lane >> 4, nl = lane & 15;

    // load x, zero-padded to 32x32
    for (int i = tid; i < NPAD * 32; i += BLOCK) {
        int m = i >> 5, k = i & 31;
        float v = (m < JN && k < in_dim) ? gload_f(x + m * in_dim + k) : 0.0f;
        s_x[i] = (half_t)v;
    }
    __syncthreads();

    // tok = x @ Wtok + b   (K padded to 32, pad rows of W are zero)
    mm_w<false, false>(s_x, 32, Wtok, btok, 32, 64, s_tok, 64, 0, nullptr, 0);
    __syncthreads();

    // qkv = tok @ Wqkv + b    [32 x 192]
    mm_w<false, false>(s_tok, 64, Wqkv, bqkv, 64, 192, s_qkv, 192, 0, nullptr, 0);
    __syncthreads();

    // vT[c][s] = V[s][c] = qkv[s][128+c]   (64 x 32)
    for (int i = tid; i < 64 * 32; i += BLOCK) {
        int c = i >> 5, s = i & 31;
        s_vT[i] = s_qkv[s * 192 + 128 + c];
    }

    // dots = Q @ K^T  (per head; K^T fragment == row-gather of K rows)
    for (int t = wave; t < heads * 4; t += NWAVES) {
        int h = t >> 2, mt = (t >> 1) & 1, nt = t & 1;
        v16h a = load_frag_lds_kv(s_qkv + mt * 16 * 192 + h * hd,      192, hd);
        v16h b = load_frag_lds_kv(s_qkv + nt * 16 * 192 + 64 + h * hd, 192, hd);
        v8f acc = {};
        acc = __builtin_amdgcn_wmma_f32_16x16x32_f16(false, a, false, b,
                                                     (short)0, acc, false, false);
        union { v8f v; float f[8]; } u; u.v = acc;
#pragma unroll
        for (int r = 0; r < 8; ++r)
            s_dots[h * 1024 + ((mt << 4) + (hi << 3) + r) * 32 + (nt << 4) + nl] = u.f[r];
    }
    __syncthreads();

    // softmax over j<22, scale = 64^-0.5; padded columns -> exactly 0
    const float scale = 0.125f;
    for (int idx = tid; idx < heads * 32; idx += BLOCK) {
        float* row = s_dots + idx * 32;
        float mx = -1e30f;
#pragma unroll
        for (int j = 0; j < JN; ++j) { float v = row[j] * scale; mx = fmaxf(mx, v); }
        float e[JN]; float sum = 0.0f;
#pragma unroll
        for (int j = 0; j < JN; ++j) { e[j] = __expf(row[j] * scale - mx); sum += e[j]; }
        float inv = 1.0f / sum;
        half_t* ar = s_attn + idx * 32;
#pragma unroll
        for (int j = 0; j < JN; ++j) ar[j] = (half_t)(e[j] * inv);
#pragma unroll
        for (int j = JN; j < 32; ++j) ar[j] = (half_t)0.0f;
    }
    __syncthreads();

    // ctx = attn @ V   (B fragment from V^T rows)
    const int nthd = hd >> 4;
    for (int t = wave; t < heads * 2 * nthd; t += NWAVES) {
        int h = t / (2 * nthd); int rem = t % (2 * nthd);
        int mt = rem / nthd, nt = rem % nthd;
        v16h a = load_frag_lds(s_attn + h * 1024 + mt * 16 * 32, 32);
        v16h b = load_frag_lds(s_vT + (h * hd + nt * 16) * 32,   32);
        v8f acc = {};
        acc = __builtin_amdgcn_wmma_f32_16x16x32_f16(false, a, false, b,
                                                     (short)0, acc, false, false);
        union { v8f v; float f[8]; } u; u.v = acc;
#pragma unroll
        for (int r = 0; r < 8; ++r)
            s_ctx[((mt << 4) + (hi << 3) + r) * 64 + h * hd + (nt << 4) + nl] = (half_t)u.f[r];
    }
    __syncthreads();

    // h = ctx @ Wo + bo
    mm_w<false, false>(s_ctx, 64, Wo, bo, 64, d_o, s_h, d_o, 0, nullptr, 0);
    __syncthreads();

    // LayerNorm per row (f32)
    if (tid < 32) {
        const half_t* row = s_h + tid * d_o;
        float m = 0.0f;
        for (int j = 0; j < d_o; ++j) m += (float)row[j];
        m /= (float)d_o;
        float v = 0.0f;
        for (int j = 0; j < d_o; ++j) { float d = (float)row[j] - m; v += d * d; }
        v /= (float)d_o;
        float inv = rsqrtf(v + 1e-5f);
        half_t* o = s_hn + tid * d_o;
        for (int j = 0; j < d_o; ++j)
            o[j] = (half_t)(((float)row[j] - m) * inv * gload_f(ln_g + j) + gload_f(ln_b + j));
    }
    __syncthreads();

    // MLP + residual -> cat slice
    mm_w<true, false>(s_hn, d_o, Wm1, bm1, d_o, 2 * d_o, s_mlp, 2 * d_o, 0, nullptr, 0);
    __syncthreads();
    mm_w<false, true>(s_mlp, 2 * d_o, Wm2, bm2, 2 * d_o, d_o, s_cat, 512, cat_ofs, s_h, d_o);
    __syncthreads();
}

// ---------------- main kernel ----------------
struct Params {
    const float *q_t, *skelA, *skelB, *quat_mean, *quat_std;
    const float *qe_tok_b, *qe_qkv_b, *qe_o_b, *qe_ln_g, *qe_ln_b, *qe_m1_b, *qe_m2_b;
    const float *se_tok_b, *se_qkv_b, *se_o_b, *se_ln_g, *se_ln_b, *se_m1_b, *se_m2_b;
    const float *emb_b, *del_W, *del_b;
    const half_t *pe;
    const half_t *qe_tok_t, *qe_qkv_t, *qe_o_t, *qe_m1_t, *qe_m2_t;
    const half_t *se_tok_t, *se_qkv_t, *se_o_t, *se_m1_t, *se_m2_t;
    const half_t *emb_t;
    float* out;
};

#define SMEM_BYTES (98 * 1024)

__global__ __launch_bounds__(BLOCK) void k_main(Params P) {
    extern __shared__ __align__(16) char smem[];
    // aliased arena (phase-ordered by barriers)
    half_t* s_cat  = (half_t*)(smem);               // 32KB persistent
    half_t* s_h    = (half_t*)(smem + 32 * 1024);   // 16KB per-encoder
    half_t* s_x    = (half_t*)(smem + 48 * 1024);   // 2KB
    half_t* s_tok  = (half_t*)(smem + 50 * 1024);   // 4KB
    half_t* s_qkv  = (half_t*)(smem + 54 * 1024);   // 12KB
    half_t* s_vT   = (half_t*)(smem + 66 * 1024);   // 4KB
    float*  s_dots = (float*) (smem + 70 * 1024);   // 16KB f32
    half_t* s_attn = (half_t*)(smem + 86 * 1024);   // 8KB
    half_t* s_ctx  = (half_t*)(smem + 94 * 1024);   // 4KB
    half_t* s_hn   = (half_t*)(smem + 48 * 1024);   // alias (x/tok/qkv dead)
    half_t* s_mlp  = (half_t*)(smem + 64 * 1024);   // alias (vT/dots/attn/ctx dead)
    half_t* s_emb  = (half_t*)(smem + 48 * 1024);   // alias, head phase
    float*  s_delta= (float*) (smem + 96 * 1024);   // head phase

    const int frame = blockIdx.x;
    const int tid = threadIdx.x;

    encoder(P.q_t + (size_t)frame * (JN * 4), 4, /*heads=*/4, /*hd=*/16, /*d_o=*/256,
            P.qe_tok_t, P.qe_tok_b, P.qe_qkv_t, P.qe_qkv_b, P.qe_o_t, P.qe_o_b,
            P.qe_ln_g, P.qe_ln_b, P.qe_m1_t, P.qe_m1_b, P.qe_m2_t, P.qe_m2_b,
            /*cat_ofs=*/0,
            s_x, s_tok, s_qkv, s_vT, s_dots, s_attn, s_ctx, s_h, s_hn, s_mlp, s_cat);

    encoder(P.skelA + (size_t)frame * (JN * 3), 3, 2, 32, 128,
            P.se_tok_t, P.se_tok_b, P.se_qkv_t, P.se_qkv_b, P.se_o_t, P.se_o_b,
            P.se_ln_g, P.se_ln_b, P.se_m1_t, P.se_m1_b, P.se_m2_t, P.se_m2_b,
            /*cat_ofs=*/256,
            s_x, s_tok, s_qkv, s_vT, s_dots, s_attn, s_ctx, s_h, s_hn, s_mlp, s_cat);

    encoder(P.skelB + (size_t)frame * (JN * 3), 3, 2, 32, 128,
            P.se_tok_t, P.se_tok_b, P.se_qkv_t, P.se_qkv_b, P.se_o_t, P.se_o_b,
            P.se_ln_g, P.se_ln_b, P.se_m1_t, P.se_m1_b, P.se_m2_t, P.se_m2_b,
            /*cat_ofs=*/384,
            s_x, s_tok, s_qkv, s_vT, s_dots, s_attn, s_ctx, s_h, s_hn, s_mlp, s_cat);

    // pos = cat * sqrt(512) + pe  (in place, f16)
    const float sq = 22.627416997969522f;
    for (int i = tid; i < NPAD * 512; i += BLOCK) {
        int m = i >> 9, c = i & 511;
        float pev = (m < JN) ? (float)(*(ghalf_t*)(P.pe + m * 512 + c)) : 0.0f;
        float v = (float)s_cat[i] * sq + pev;
        s_cat[i] = (half_t)v;
    }
    __syncthreads();

    // embed = relu(pos @ emb_W + emb_b)   [32 x 128]
    mm_w<true, false>(s_cat, 512, P.emb_t, P.emb_b, 512, 128, s_emb, 128, 0, nullptr, 0);
    __syncthreads();

    // delta = embed @ del_W + del_b ; affine with quat std/mean
    if (tid < JN * 4) {
        int m = tid >> 2, c = tid & 3;
        float acc = gload_f(P.del_b + c);
        for (int k = 0; k < 128; ++k)
            acc += (float)s_emb[m * 128 + k] * gload_f(P.del_W + k * 4 + c);
        acc = acc * gload_f(P.quat_std + m * 4 + c) + gload_f(P.quat_mean + m * 4 + c);
        s_delta[tid] = acc;
    }
    __syncthreads();

    // normalize + write out
    if (tid < JN) {
        float a = s_delta[tid * 4 + 0], b = s_delta[tid * 4 + 1];
        float c = s_delta[tid * 4 + 2], d = s_delta[tid * 4 + 3];
        float inv = rsqrtf(a * a + b * b + c * c + d * d);
        float* o = P.out + ((size_t)frame * JN + tid) * 4;
        o[0] = a * inv; o[1] = b * inv; o[2] = c * inv; o[3] = d * inv;
    }
}

// ---------------- prep kernels ----------------
// sinusoidal PE table, f16, [22 x 512]
__global__ void k_pe(half_t* pe) {
    int i = blockIdx.x * blockDim.x + threadIdx.x;
    if (i >= JN * 512) return;
    int p = i >> 9, c = i & 511;
    float div = __expf((float)(c & ~1) * (-9.210340371976184f / 512.0f));
    float ang = (float)p * div;
    pe[i] = (half_t)((c & 1) ? __cosf(ang) : __sinf(ang));
}

// pack fp32 W[Ksrc x N] -> f16 tiles (Kpad/32 x N/16 tiles, 512 halves each,
// lane-major per the WMMA B-operand striping); zero-fill K padding.
__global__ void k_pack(const float* __restrict__ W, half_t* __restrict__ dst,
                       int Ksrc, int N, int Kpad) {
    int ntk = Kpad >> 5;
    int ntn = N >> 4;
    int total = ntk * ntn * 512;
    int idx = blockIdx.x * blockDim.x + threadIdx.x;
    if (idx >= total) return;
    int tile = idx >> 9;
    int r    = idx & 511;
    int lane = r >> 4;
    int j    = r & 15;
    int nt = tile / ntk;
    int kt = tile % ntk;
    int hihalf = lane >> 4;
    int nl = lane & 15;
    int kl = (j < 8) ? ((hihalf << 3) + j) : (16 + (hihalf << 3) + (j - 8));
    int k = (kt << 5) + kl;
    int n = (nt << 4) + nl;
    float v = (k < Ksrc) ? gload_f(W + (size_t)k * N + n) : 0.0f;
    dst[idx] = (half_t)v;
}

// ---------------- host launch ----------------
extern "C" void kernel_launch(void* const* d_in, const int* in_sizes, int n_in,
                              void* d_out, int out_size, void* d_ws, size_t ws_size,
                              hipStream_t stream) {
    half_t* ws = (half_t*)d_ws;
    size_t off = 0;

    half_t* pe = ws + off; off += (size_t)JN * 512;
    {
        int total = JN * 512;
        k_pe<<<(total + 255) / 256, 256, 0, stream>>>(pe);
    }

    auto pack = [&](const void* W, int Ksrc, int N, int Kpad) -> half_t* {
        half_t* dst = ws + off;
        int ntk = Kpad / 32, ntn = N / 16;
        int total = ntk * ntn * 512;
        k_pack<<<(total + 255) / 256, 256, 0, stream>>>((const float*)W, dst, Ksrc, N, Kpad);
        off += (size_t)total;
        return dst;
    };

    half_t* qe_tok_t = pack(d_in[5],  4,   64,  32);
    half_t* qe_qkv_t = pack(d_in[7],  64,  192, 64);
    half_t* qe_o_t   = pack(d_in[9],  64,  256, 64);
    half_t* qe_m1_t  = pack(d_in[13], 256, 512, 256);
    half_t* qe_m2_t  = pack(d_in[15], 512, 256, 512);
    half_t* se_tok_t = pack(d_in[17], 3,   64,  32);
    half_t* se_qkv_t = pack(d_in[19], 64,  192, 64);
    half_t* se_o_t   = pack(d_in[21], 64,  128, 64);
    half_t* se_m1_t  = pack(d_in[25], 128, 256, 128);
    half_t* se_m2_t  = pack(d_in[27], 256, 128, 256);
    half_t* emb_t    = pack(d_in[29], 512, 128, 512);

    Params P;
    P.q_t       = (const float*)d_in[0];
    P.skelA     = (const float*)d_in[1];
    P.skelB     = (const float*)d_in[2];
    P.quat_mean = (const float*)d_in[3];
    P.quat_std  = (const float*)d_in[4];
    P.qe_tok_b  = (const float*)d_in[6];
    P.qe_qkv_b  = (const float*)d_in[8];
    P.qe_o_b    = (const float*)d_in[10];
    P.qe_ln_g   = (const float*)d_in[11];
    P.qe_ln_b   = (const float*)d_in[12];
    P.qe_m1_b   = (const float*)d_in[14];
    P.qe_m2_b   = (const float*)d_in[16];
    P.se_tok_b  = (const float*)d_in[18];
    P.se_qkv_b  = (const float*)d_in[20];
    P.se_o_b    = (const float*)d_in[22];
    P.se_ln_g   = (const float*)d_in[23];
    P.se_ln_b   = (const float*)d_in[24];
    P.se_m1_b   = (const float*)d_in[26];
    P.se_m2_b   = (const float*)d_in[28];
    P.emb_b     = (const float*)d_in[30];
    P.del_W     = (const float*)d_in[31];
    P.del_b     = (const float*)d_in[32];
    P.pe = pe;
    P.qe_tok_t = qe_tok_t; P.qe_qkv_t = qe_qkv_t; P.qe_o_t = qe_o_t;
    P.qe_m1_t = qe_m1_t;   P.qe_m2_t = qe_m2_t;
    P.se_tok_t = se_tok_t; P.se_qkv_t = se_qkv_t; P.se_o_t = se_o_t;
    P.se_m1_t = se_m1_t;   P.se_m2_t = se_m2_t;
    P.emb_t = emb_t;
    P.out = (float*)d_out;

    (void)hipFuncSetAttribute((const void*)k_main,
                              hipFuncAttributeMaxDynamicSharedMemorySize,
                              SMEM_BYTES);

    int frames = in_sizes[0] / (JN * 4);
    k_main<<<frames, BLOCK, SMEM_BYTES, stream>>>(P);
}
